// TFEfficientFormerMeta3D_47253230191072
// MI455X (gfx1250) — matrix operand order repaired and checked by reading, not compiled, not run on hardware
//
#include <hip/hip_runtime.h>
#include <math.h>

// ---------------- problem constants ----------------
#define B_      512
#define S_      49
#define D_      448
#define H_      8
#define KD_     32
#define EKD_    128          // KD*RATIO
#define QKVO_   1536         // H*(2*KD+EKD)
#define MLPH_   1792         // 4*D
#define TOK_    (B_ * S_)    // 25088 = 392*64
#define VROW_   64           // padded V row (s dimension 49 -> 64)

typedef __bf16 bf16;
typedef __bf16 v16bf __attribute__((ext_vector_type(16)));
typedef float  v8f   __attribute__((ext_vector_type(8)));

// ---------------------------------------------------------------------------
// transpose + cast weights: in[K][N] f32 -> out[N][K] bf16 (32x32 LDS tiles)
// ---------------------------------------------------------------------------
__global__ __launch_bounds__(256)
void transpose_cast(const float* __restrict__ in, bf16* __restrict__ out, int K, int N) {
  __shared__ float t[32][33];
  const int k0 = blockIdx.y * 32, n0 = blockIdx.x * 32;
  const int tx = threadIdx.x, ty = threadIdx.y;          // (32, 8)
  #pragma unroll
  for (int i = 0; i < 32; i += 8) {
    const int k = k0 + ty + i, n = n0 + tx;
    t[ty + i][tx] = (k < K && n < N) ? in[(size_t)k * N + n] : 0.0f;
  }
  __syncthreads();
  #pragma unroll
  for (int i = 0; i < 32; i += 8) {
    const int n = n0 + ty + i, k = k0 + tx;
    if (n < N && k < K) out[(size_t)n * K + k] = (bf16)t[tx][ty + i];
  }
}

// ---------------------------------------------------------------------------
// dense attention bias: db[H][64][64] = biases[h, idxs[q,k]], mask baked in.
// Removes the per-element dependent gather from the attention inner loop.
// ---------------------------------------------------------------------------
__global__ void build_bias(const float* __restrict__ biases, const int* __restrict__ bidx,
                           int nOff, float* __restrict__ db) {
  const int i = blockIdx.x * blockDim.x + threadIdx.x;
  if (i >= H_ * 64 * 64) return;
  const int h = i >> 12, row = (i >> 6) & 63, col = i & 63;
  float v;
  if (row < S_ && col < S_) v = biases[(size_t)h * nOff + bidx[row * S_ + col]];
  else v = (col >= S_) ? -1e30f : 0.0f;   // mask padded key columns
  db[i] = v;
}

// ---------------------------------------------------------------------------
// LayerNorm over D_=448, one block (256 threads) per token -> bf16
// ---------------------------------------------------------------------------
__global__ __launch_bounds__(256)
void ln_to_bf16(const float* __restrict__ x, const float* __restrict__ g,
                const float* __restrict__ b, bf16* __restrict__ out) {
  const int row = blockIdx.x;
  const float* xr = x + (size_t)row * D_;
  const int t = threadIdx.x;

  float x0 = xr[t];
  float x1 = (t + 256 < D_) ? xr[t + 256] : 0.0f;
  float s  = x0 + x1;
  float sq = x0 * x0 + x1 * x1;
  #pragma unroll
  for (int m = 16; m >= 1; m >>= 1) { s += __shfl_xor(s, m); sq += __shfl_xor(sq, m); }
  __shared__ float sh[2][8];
  const int lane = t & 31, w = t >> 5;
  if (lane == 0) { sh[0][w] = s; sh[1][w] = sq; }
  __syncthreads();
  if (t == 0) {
    float S = 0.f, Q = 0.f;
    #pragma unroll
    for (int i = 0; i < 8; ++i) { S += sh[0][i]; Q += sh[1][i]; }
    sh[0][0] = S; sh[1][0] = Q;
  }
  __syncthreads();
  const float mu   = sh[0][0] * (1.0f / D_);
  const float var  = sh[1][0] * (1.0f / D_) - mu * mu;
  const float rstd = rsqrtf(var + 1e-5f);
  out[(size_t)row * D_ + t] = (bf16)((x0 - mu) * rstd * g[t] + b[t]);
  if (t + 256 < D_)
    out[(size_t)row * D_ + t + 256] = (bf16)((x1 - mu) * rstd * g[t + 256] + b[t + 256]);
}

// ---------------------------------------------------------------------------
// WMMA bf16 GEMM: C = A(MxK, row-major) * W(KxN given as Wt[N][K]) + epilogue.
// Block tile 64(M) x 128(N), 8 waves (2x4), 32x32 per wave.
// Double-buffered LDS with register prefetch; all staging uint4/b128;
// both A and B fragment LDS reads are K-contiguous -> ds_load_b128 pairs.
// ---------------------------------------------------------------------------
enum { EPI_QKV = 0, EPI_RESID = 1, EPI_GELU = 2 };

template <int EPI>
__global__ __launch_bounds__(256)
void gemm_wmma(const bf16* __restrict__ A, const bf16* __restrict__ Wt,
               int M, int N, int K,
               const float* __restrict__ bias,
               const float* __restrict__ scl,
               const float* __restrict__ resid,
               float* __restrict__ outF,
               bf16* __restrict__ outQ, bf16* __restrict__ outK2, bf16* __restrict__ outV)
{
  __shared__ __align__(16) bf16 As [2][64][40];    // [m][k], 80B rows
  __shared__ __align__(16) bf16 Bst[2][128][40];   // [n][k], 80B rows

  const int tid    = threadIdx.x;
  const int lane   = tid & 31;
  const int wv     = tid >> 5;
  const int wm     = wv >> 2;          // 0..1
  const int wn     = wv & 3;           // 0..3
  const int halfId = lane >> 4;
  const int l15    = lane & 15;
  const int mBase  = blockIdx.y * 64;
  const int nBase  = blockIdx.x * 128;

  // staging assignment
  const int ar = tid >> 2, acb = (tid & 3) * 8;    // A: 64 rows x 32k, 8 bf16/thr
  const int bn = tid >> 1, bkc = (tid & 1) * 16;   // B: 128 rows x 32k, 16 bf16/thr
  const bool aValid = (mBase + ar) < M;
  const bool bValid = (nBase + bn) < N;
  const bf16* aPtr = A  + (size_t)(mBase + ar) * K + acb;
  const bf16* bPtr = Wt + (size_t)(nBase + bn) * K + bkc;
  const uint4 z4 = make_uint4(0u, 0u, 0u, 0u);

  uint4 aReg, bReg0, bReg1;
  auto loadRegs = [&](int k0) {
    aReg = aValid ? *reinterpret_cast<const uint4*>(aPtr + k0) : z4;
    if (bValid) {
      bReg0 = *reinterpret_cast<const uint4*>(bPtr + k0);
      bReg1 = *reinterpret_cast<const uint4*>(bPtr + k0 + 8);
    } else { bReg0 = z4; bReg1 = z4; }
  };
  auto storeRegs = [&](int buf) {
    *reinterpret_cast<uint4*>(&As [buf][ar][acb])     = aReg;
    *reinterpret_cast<uint4*>(&Bst[buf][bn][bkc])     = bReg0;
    *reinterpret_cast<uint4*>(&Bst[buf][bn][bkc + 8]) = bReg1;
  };

  v8f c[2][2] = {};
  loadRegs(0);
  storeRegs(0);
  int cur = 0;

  for (int k0 = 0; k0 < K; k0 += 32) {
    const bool more = (k0 + 32) < K;
    if (more) loadRegs(k0 + 32);     // prefetch next tile while computing this one
    __syncthreads();

    v16bf afrag[2], bfrag[2];
    #pragma unroll
    for (int mt = 0; mt < 2; ++mt) {
      const int m = wm * 32 + mt * 16 + l15;
      #pragma unroll
      for (int j = 0; j < 16; ++j) {           // A: K-contiguous, k = 16*half + j (matches B)
        const int kk = halfId * 16 + j;
        afrag[mt][j] = As[cur][m][kk];
      }
    }
    #pragma unroll
    for (int nt = 0; nt < 2; ++nt) {
      const int n = wn * 32 + nt * 16 + l15;
      #pragma unroll
      for (int j = 0; j < 16; ++j)             // K-contiguous -> b128 pair
        bfrag[nt][j] = Bst[cur][n][halfId * 16 + j];
    }
    #pragma unroll
    for (int mt = 0; mt < 2; ++mt)
      #pragma unroll
      for (int nt = 0; nt < 2; ++nt)
        c[mt][nt] = __builtin_amdgcn_wmma_f32_16x16x32_bf16(
            false, afrag[mt], false, bfrag[nt], (short)0, c[mt][nt], false, false);

    __syncthreads();
    if (more) storeRegs(cur ^ 1);
    cur ^= 1;
  }

  // epilogue: VGPR r holds row (r + 8*halfId), lane&15 is column
  #pragma unroll
  for (int mt = 0; mt < 2; ++mt) {
    #pragma unroll
    for (int nt = 0; nt < 2; ++nt) {
      #pragma unroll
      for (int r = 0; r < 8; ++r) {
        const int m = mBase + wm * 32 + mt * 16 + r + halfId * 8;
        const int n = nBase + wn * 32 + nt * 16 + l15;
        if (m < M && n < N) {
          float acc = c[mt][nt][r] + bias[n];
          if constexpr (EPI == EPI_QKV) {
            const int bb = m / S_;
            const int ss = m - bb * S_;
            const int hh = n / 192;
            const int rc = n - hh * 192;
            const size_t th = ((size_t)bb * H_ + hh) * S_ + ss;
            if (rc < KD_)          outQ [th * KD_ + rc]         = (bf16)acc;
            else if (rc < 2 * KD_) outK2[th * KD_ + (rc - KD_)] = (bf16)acc;
            else { // V transposed: [bh][d][VROW_] so attention B-frags are contiguous
              const size_t dd = rc - 2 * KD_;
              outV[(((size_t)bb * H_ + hh) * EKD_ + dd) * VROW_ + ss] = (bf16)acc;
            }
          } else if constexpr (EPI == EPI_RESID) {
            const size_t idx = (size_t)m * N + n;
            outF[idx] = resid[idx] + scl[n] * acc;
          } else { // EPI_GELU exact
            const float gl = 0.5f * acc * (1.0f + erff(acc * 0.70710678118654752f));
            outQ[(size_t)m * N + n] = (bf16)gl;
          }
        }
      }
    }
  }
}

// ---------------------------------------------------------------------------
// Attention: one block (128 thr = 4 waves) per (b,h).
// scores = q@k^T via WMMA; dense precomputed bias add (coalesced, branch-free);
// register softmax via half-wave __shfl_xor; probs(bf16) @ v via WMMA.
// ---------------------------------------------------------------------------
__global__ __launch_bounds__(128)
void attn_wmma(const bf16* __restrict__ qb, const bf16* __restrict__ kb,
               const bf16* __restrict__ vt,     // [bh][d][VROW_]
               const float* __restrict__ db,    // [H][64][64] dense bias (masked)
               bf16* __restrict__ ctx)
{
  const int bh = blockIdx.x;
  const int b  = bh / H_;
  const int h  = bh - b * H_;

  __shared__ __align__(16) bf16 qs [64][40];
  __shared__ __align__(16) bf16 ks [64][40];
  __shared__ __align__(16) bf16 vst[128][72];   // [d][s], 144B rows
  __shared__ __align__(16) bf16 ps [64][72];    // probs [q][kcol]

  const int tid = threadIdx.x;
  { // stage q,k (49x32 zero-padded to 64x32), uint4 copies
    const int r = tid >> 1, cb = (tid & 1) * 16;
    if (r < S_) {
      const bf16* qsrc = qb + ((size_t)bh * S_ + r) * KD_ + cb;
      const bf16* ksrc = kb + ((size_t)bh * S_ + r) * KD_ + cb;
      *reinterpret_cast<uint4*>(&qs[r][cb])     = *reinterpret_cast<const uint4*>(qsrc);
      *reinterpret_cast<uint4*>(&qs[r][cb + 8]) = *reinterpret_cast<const uint4*>(qsrc + 8);
      *reinterpret_cast<uint4*>(&ks[r][cb])     = *reinterpret_cast<const uint4*>(ksrc);
      *reinterpret_cast<uint4*>(&ks[r][cb + 8]) = *reinterpret_cast<const uint4*>(ksrc + 8);
    } else {
      const uint4 z4 = make_uint4(0u, 0u, 0u, 0u);
      *reinterpret_cast<uint4*>(&qs[r][cb])     = z4;
      *reinterpret_cast<uint4*>(&qs[r][cb + 8]) = z4;
      *reinterpret_cast<uint4*>(&ks[r][cb])     = z4;
      *reinterpret_cast<uint4*>(&ks[r][cb + 8]) = z4;
    }
  }
  { // stage v^T: one 64-wide row per thread (d = tid), vectorized
    const bf16* vsrc = vt + ((size_t)bh * EKD_ + tid) * VROW_;
    #pragma unroll
    for (int cch = 0; cch < 6; ++cch)            // s = 0..47
      *reinterpret_cast<uint4*>(&vst[tid][cch * 8]) =
          *reinterpret_cast<const uint4*>(vsrc + cch * 8);
    vst[tid][48] = vsrc[48];
    #pragma unroll
    for (int j = 49; j < 64; ++j) vst[tid][j] = (bf16)0.0f;
  }
  __syncthreads();

  const int lane = tid & 31, wv = tid >> 5;
  const int halfId = lane >> 4, l15 = lane & 15;

  // ---- scores: wave wv owns query rows [16*wv, 16*wv+16) ----
  v16bf aq;
  {
    const int m = wv * 16 + l15;
    #pragma unroll
    for (int j = 0; j < 16; ++j) {
      const int kk = halfId * 16 + j;
      aq[j] = qs[m][kk];
    }
  }
  v8f sc[4] = {};
  #pragma unroll
  for (int nt = 0; nt < 4; ++nt) {
    v16bf bk;
    const int n = nt * 16 + l15;
    #pragma unroll
    for (int j = 0; j < 16; ++j) bk[j] = ks[n][halfId * 16 + j];
    sc[nt] = __builtin_amdgcn_wmma_f32_16x16x32_bf16(
        false, aq, false, bk, (short)0, sc[nt], false, false);
  }

  // ---- scale + dense bias + softmax in registers (no branches) ----
  float pv[4][8];
  #pragma unroll
  for (int nt = 0; nt < 4; ++nt)
    #pragma unroll
    for (int r = 0; r < 8; ++r) {
      const int row = wv * 16 + r + halfId * 8;
      const int col = nt * 16 + l15;
      pv[nt][r] = sc[nt][r] * 0.17677669529663687f
                + db[(((size_t)h * 64) + row) * 64 + col];
    }
  #pragma unroll
  for (int r = 0; r < 8; ++r) {
    float mx = -1e30f;
    #pragma unroll
    for (int nt = 0; nt < 4; ++nt) {
      float v = pv[nt][r];
      #pragma unroll
      for (int m = 8; m >= 1; m >>= 1) v = fmaxf(v, __shfl_xor(v, m));
      mx = fmaxf(mx, v);
    }
    float sum = 0.0f;
    #pragma unroll
    for (int nt = 0; nt < 4; ++nt) {
      float e = __expf(pv[nt][r] - mx);
      pv[nt][r] = e;
      #pragma unroll
      for (int m = 8; m >= 1; m >>= 1) e += __shfl_xor(e, m);
      sum += e;
    }
    const float inv = 1.0f / sum;
    #pragma unroll
    for (int nt = 0; nt < 4; ++nt) pv[nt][r] *= inv;
  }
  #pragma unroll
  for (int nt = 0; nt < 4; ++nt)
    #pragma unroll
    for (int r = 0; r < 8; ++r)
      ps[wv * 16 + r + halfId * 8][nt * 16 + l15] = (bf16)pv[nt][r];
  __syncthreads();

  // ---- ctx = probs(64x64) @ v(64x128); B-frags contiguous via vst[d][s] ----
  v8f cacc[8] = {};
  #pragma unroll
  for (int kc = 0; kc < 2; ++kc) {
    v16bf ap;
    const int m = wv * 16 + l15;
    #pragma unroll
    for (int j = 0; j < 16; ++j) {
      const int kk = kc * 32 + halfId * 16 + j;
      ap[j] = ps[m][kk];
    }
    #pragma unroll
    for (int nt = 0; nt < 8; ++nt) {
      v16bf bvv;
      const int n = nt * 16 + l15;
      #pragma unroll
      for (int j = 0; j < 16; ++j) bvv[j] = vst[n][kc * 32 + halfId * 16 + j];
      cacc[nt] = __builtin_amdgcn_wmma_f32_16x16x32_bf16(
          false, ap, false, bvv, (short)0, cacc[nt], false, false);
    }
  }
  #pragma unroll
  for (int nt = 0; nt < 8; ++nt)
    #pragma unroll
    for (int r = 0; r < 8; ++r) {
      const int row = wv * 16 + r + halfId * 8;
      const int col = nt * 16 + l15;
      if (row < S_)
        ctx[((size_t)b * S_ + row) * (H_ * EKD_) + (size_t)h * EKD_ + col] = (bf16)cacc[nt][r];
    }
}

// ---------------------------------------------------------------------------
// host-side orchestration
// ---------------------------------------------------------------------------
extern "C" void kernel_launch(void* const* d_in, const int* in_sizes, int n_in,
                              void* d_out, int out_size, void* d_ws, size_t ws_size,
                              hipStream_t stream) {
  const float* x       = (const float*)d_in[0];
  const float* ln1_g   = (const float*)d_in[1];
  const float* ln1_b   = (const float*)d_in[2];
  const float* qkv_w   = (const float*)d_in[3];
  const float* qkv_b   = (const float*)d_in[4];
  const float* att_bias= (const float*)d_in[5];
  const float* proj_w  = (const float*)d_in[6];
  const float* proj_b  = (const float*)d_in[7];
  const float* ln2_g   = (const float*)d_in[8];
  const float* ln2_b   = (const float*)d_in[9];
  const float* mlp_w1  = (const float*)d_in[10];
  const float* mlp_b1  = (const float*)d_in[11];
  const float* mlp_w2  = (const float*)d_in[12];
  const float* mlp_b2  = (const float*)d_in[13];
  const float* ls1     = (const float*)d_in[14];
  const float* ls2     = (const float*)d_in[15];
  const int*   bidx    = (const int*)d_in[16];
  const int    nOff    = in_sizes[5] / H_;
  float* out = (float*)d_out;
  char* ws = (char*)d_ws;

  const size_t SZ_WQKV = (size_t)QKVO_ * D_ * 2;
  const size_t SZ_WPRJ = (size_t)D_ * (H_ * EKD_) * 2;
  const size_t SZ_W1   = (size_t)MLPH_ * D_ * 2;
  const size_t SZ_W2   = (size_t)D_ * MLPH_ * 2;
  const size_t SZ_DB   = (size_t)H_ * 64 * 64 * 4;
  const size_t SZ_X1   = (size_t)TOK_ * D_ * 4;
  const size_t SZ_H    = (size_t)TOK_ * D_ * 2;
  const size_t SZ_Q    = (size_t)B_ * H_ * S_ * KD_ * 2;
  const size_t SZ_V    = (size_t)B_ * H_ * EKD_ * VROW_ * 2;

  bf16*  wqkvT = (bf16*)(ws);
  bf16*  wprjT = (bf16*)(ws + SZ_WQKV);
  bf16*  w1T   = (bf16*)(ws + SZ_WQKV + SZ_WPRJ);
  bf16*  w2T   = (bf16*)(ws + SZ_WQKV + SZ_WPRJ + SZ_W1);
  float* db    = (float*)(ws + SZ_WQKV + SZ_WPRJ + SZ_W1 + SZ_W2);
  float* x1    = (float*)(ws + SZ_WQKV + SZ_WPRJ + SZ_W1 + SZ_W2 + SZ_DB);
  bf16*  hbf   = (bf16*) (ws + SZ_WQKV + SZ_WPRJ + SZ_W1 + SZ_W2 + SZ_DB + SZ_X1);
  char*  big   =          ws + SZ_WQKV + SZ_WPRJ + SZ_W1 + SZ_W2 + SZ_DB + SZ_X1 + SZ_H;
  bf16*  qbuf  = (bf16*)(big);
  bf16*  kbuf  = (bf16*)(big + SZ_Q);
  bf16*  vbuf  = (bf16*)(big + 2 * SZ_Q);
  bf16*  ctx   = (bf16*)(big + 2 * SZ_Q + SZ_V);
  bf16*  mh    = (bf16*)(big);               // aliases q/k/v/ctx (dead by then)

  // 1) weights -> bf16, transposed to [N][K] for K-contiguous B fragments
  {
    dim3 blk(32, 8);
    transpose_cast<<<dim3(QKVO_ / 32, D_ / 32), blk, 0, stream>>>(qkv_w, wqkvT, D_, QKVO_);
    transpose_cast<<<dim3(D_ / 32, (H_ * EKD_) / 32), blk, 0, stream>>>(proj_w, wprjT, H_ * EKD_, D_);
    transpose_cast<<<dim3(MLPH_ / 32, D_ / 32), blk, 0, stream>>>(mlp_w1, w1T, D_, MLPH_);
    transpose_cast<<<dim3(D_ / 32, MLPH_ / 32), blk, 0, stream>>>(mlp_w2, w2T, MLPH_, D_);
  }
  // 2) dense masked attention bias [H][64][64]
  build_bias<<<(H_ * 64 * 64 + 255) / 256, 256, 0, stream>>>(att_bias, bidx, nOff, db);

  // 3) LN1 -> h (bf16)
  ln_to_bf16<<<TOK_, 256, 0, stream>>>(x, ln1_g, ln1_b, hbf);

  // 4) QKV GEMM (25088 x 448 x 1536) with q/k/v^T scatter epilogue
  gemm_wmma<EPI_QKV><<<dim3(QKVO_ / 128, TOK_ / 64), 256, 0, stream>>>(
      hbf, wqkvT, TOK_, QKVO_, D_, qkv_b, nullptr, nullptr, nullptr, qbuf, kbuf, vbuf);

  // 5) attention per (b,h)
  attn_wmma<<<B_ * H_, 128, 0, stream>>>(qbuf, kbuf, vbuf, db, ctx);

  // 6) proj GEMM (25088 x 1024 x 448) + residual*ls1 -> x1 (f32)
  gemm_wmma<EPI_RESID><<<dim3((D_ + 127) / 128, TOK_ / 64), 256, 0, stream>>>(
      ctx, wprjT, TOK_, D_, H_ * EKD_, proj_b, ls1, x, x1, nullptr, nullptr, nullptr);

  // 7) LN2 -> h2 (bf16, reuses h buffer)
  ln_to_bf16<<<TOK_, 256, 0, stream>>>(x1, ln2_g, ln2_b, hbf);

  // 8) mlp1 GEMM (25088 x 448 x 1792) + exact GELU -> mh (bf16)
  gemm_wmma<EPI_GELU><<<dim3(MLPH_ / 128, TOK_ / 64), 256, 0, stream>>>(
      hbf, w1T, TOK_, MLPH_, D_, mlp_b1, nullptr, nullptr, nullptr, mh, nullptr, nullptr);

  // 9) mlp2 GEMM (25088 x 1792 x 448) + residual*ls2 -> out (f32)
  gemm_wmma<EPI_RESID><<<dim3((D_ + 127) / 128, TOK_ / 64), 256, 0, stream>>>(
      mh, w2T, TOK_, D_, MLPH_, mlp_b2, ls2, x1, out, nullptr, nullptr, nullptr);

  (void)n_in; (void)out_size; (void)ws_size;
}
